// WindowAttention2DTime_7799660609902
// MI455X (gfx1250) — matrix operand order repaired and checked by reading, not compiled, not run on hardware
//
#include <hip/hip_runtime.h>

typedef __attribute__((ext_vector_type(16))) __bf16 v16bf;
typedef __attribute__((ext_vector_type(8)))  __bf16 v8bf;
typedef __attribute__((ext_vector_type(8)))  float  v8f;

union V16 { v16bf v; v8bf h[2]; };

#define N_TOK 64
#define HEADS 16
#define HD_   32
#define CDIM  512
#define K3    1536
#define NBATCH 1024
#define LDA   40            // padded LDS row stride (bf16 units): 80B, bank-spread, 16B aligned

// ---- CDNA5 async global->LDS copy (16B per lane), tracked by ASYNCcnt ----
__device__ __forceinline__ void wa2_async_b128(const void* gsrc, void* ldst) {
  unsigned lds = (unsigned)(size_t)ldst;                 // low 32 bits of flat LDS ptr = LDS offset
  unsigned long long g = (unsigned long long)(size_t)gsrc;
  asm volatile("global_load_async_to_lds_b128 %0, %1, off"
               :: "v"(lds), "v"(g) : "memory");
}

// ---------------- f32 -> bf16 elementwise convert ----------------
__global__ __launch_bounds__(256)
void wa2_cvt_bf16(const float* __restrict__ in, __bf16* __restrict__ out, int n) {
  int i = blockIdx.x * 256 + threadIdx.x;
  if (i < n) out[i] = (__bf16)in[i];
}

// ---------------- QKV projection: Out[m, j] = bf16( X[m,:] . W[j,:] + b[j] ) ----------------
// block = 128 threads (4 waves); tile M=128 x N=64; each wave owns a 32x64 strip (8 WMMA/K-step).
// K staged via async global->LDS, double buffered; batched ds_loads feed back-to-back WMMAs.
__global__ __launch_bounds__(128)
void wa2_qkv_gemm(const __bf16* __restrict__ X, const __bf16* __restrict__ W,
                  const float* __restrict__ bias, __bf16* __restrict__ Out) {
  const int tid  = threadIdx.x;
  const int wave = tid >> 5;
  const int lane = tid & 31;
  const int hh   = lane >> 4;      // half-wave: 0 or 1
  const int l16  = lane & 15;
  const int mb = blockIdx.x * 128;
  const int n0 = blockIdx.y * 64;

  __shared__ __align__(16) __bf16 As[2][128 * LDA];  // 128x32 slab, padded rows
  __shared__ __align__(16) __bf16 Bs[2][64 * LDA];   //  64x32 slab

  // stage one K-step: A 8KB + B 4KB = 768 x 16B chunks, 6 per thread
  auto stage = [&](int buf, int k0) {
#pragma unroll
    for (int i = 0; i < 6; ++i) {
      int c = tid + 128 * i;
      if (i < 4) {                               // A chunks 0..511
        int row = c >> 2;
        int col = (c & 3) * 8;
        wa2_async_b128(X + (size_t)(mb + row) * CDIM + k0 + col, &As[buf][row * LDA + col]);
      } else {                                   // B chunks 0..255
        int cc  = c - 512;
        int row = cc >> 2;
        int col = (cc & 3) * 8;
        wa2_async_b128(W + (size_t)(n0 + row) * CDIM + k0 + col, &Bs[buf][row * LDA + col]);
      }
    }
  };

  v8f acc[2][4] = {};
  auto kstep = [&](int cur, int ks) {            // cur is a compile-time constant at call sites
    if (ks + 1 < 16) {
      stage(cur ^ 1, (ks + 1) * 32);
      asm volatile("s_wait_asynccnt 0x6" ::: "memory");   // older stage done (6 newer in flight)
    } else {
      asm volatile("s_wait_asynccnt 0x0" ::: "memory");
    }
    __syncthreads();                                      // all waves' copies for `cur` visible

    V16 a[2], bt[4];                                      // batch ALL tile loads first
#pragma unroll
    for (int mt = 0; mt < 2; ++mt) {                      // A 16x32: K = 16*hh+i
      const __bf16* ap = &As[cur][(wave * 32 + mt * 16 + l16) * LDA + 16 * hh];
      a[mt].h[0] = *(const v8bf*)(ap);
      a[mt].h[1] = *(const v8bf*)(ap + 8);
    }
#pragma unroll
    for (int t = 0; t < 4; ++t) {                         // B 32x16: lane col = t*16+l16, K = 16*hh+i
      const __bf16* bp = &Bs[cur][(t * 16 + l16) * LDA + 16 * hh];
      bt[t].h[0] = *(const v8bf*)(bp);
      bt[t].h[1] = *(const v8bf*)(bp + 8);
    }
#pragma unroll
    for (int mt = 0; mt < 2; ++mt)                        // 8 WMMAs back-to-back
#pragma unroll
      for (int t = 0; t < 4; ++t)
        acc[mt][t] = __builtin_amdgcn_wmma_f32_16x16x32_bf16(false, a[mt].v, false, bt[t].v,
                                                             (short)0, acc[mt][t], false, false);
    __syncthreads();                                      // done reading `cur` before refill
  };

  stage(0, 0);
  for (int ks = 0; ks < 16; ks += 2) {                    // manual 2x unroll: buffer idx is constant
    kstep(0, ks);
    kstep(1, ks + 1);
  }

#pragma unroll
  for (int mt = 0; mt < 2; ++mt)
#pragma unroll
    for (int t = 0; t < 4; ++t) {
      int col = n0 + t * 16 + l16;
      float bv = bias[col];
#pragma unroll
      for (int j = 0; j < 8; ++j) {                       // C: row = j + 8*hh, col = l16
        int row = mb + wave * 32 + mt * 16 + 8 * hh + j;
        Out[(size_t)row * K3 + col] = (__bf16)(acc[mt][t][j] + bv);
      }
    }
}

// ---------------- CPB MLP: tab[p][h], p in [0,225) ----------------
__device__ __forceinline__ float wa2_cpb_coord(int i) {
  float v = (float)(i - 7) * (8.0f / 7.0f);
  float s = (v > 0.f) ? 1.f : ((v < 0.f) ? -1.f : 0.f);
  return s * log2f(fabsf(v) + 1.0f) * (1.0f / 3.0f);     // /log2(8)
}

__global__ __launch_bounds__(16)
void wa2_cpb(const float* __restrict__ w1, const float* __restrict__ b1,
             const float* __restrict__ w2, float* __restrict__ tab) {
  int p = blockIdx.x;          // 0..224
  int h = threadIdx.x;         // 0..15
  float fh = wa2_cpb_coord(p / 15);
  float fw = wa2_cpb_coord(p % 15);
  float sum = 0.f;
  for (int j = 0; j < 512; ++j) {
    float hid = fmaf(w1[2 * j], fh, fmaf(w1[2 * j + 1], fw, b1[j]));
    hid = fmaxf(hid, 0.f);
    sum = fmaf(hid, w2[h * 512 + j], sum);
  }
  tab[p * HEADS + h] = sum;
}

// ---------------- bias[h][i][j] = 16*sigmoid(tab[rpi(i,j)][h]) ----------------
__global__ __launch_bounds__(256)
void wa2_bias(const float* __restrict__ tab, float* __restrict__ bias) {
  int idx = blockIdx.x * 256 + threadIdx.x;
  if (idx >= HEADS * N_TOK * N_TOK) return;
  int h = idx >> 12;
  int i = (idx >> 6) & 63;
  int j = idx & 63;
  int p = (((i >> 3) - (j >> 3)) + 7) * 15 + (((i & 7) - (j & 7)) + 7);
  float v = tab[p * HEADS + h];
  bias[idx] = 16.0f / (1.0f + __expf(-v));
}

// ---------------- attention: one block per (window b, head h), 4 waves ----------------
__global__ __launch_bounds__(128)
void wa2_attn(const __bf16* __restrict__ QKV, const float* __restrict__ mask,
              const float* __restrict__ bias, const float* __restrict__ logit_scale,
              float* __restrict__ out) {
  const int b = blockIdx.x;
  const int h = blockIdx.y;
  const int wave = threadIdx.x >> 5;
  const int lane = threadIdx.x & 31;
  const int hh   = lane >> 4;
  const int l16  = lane & 15;

  __shared__ __align__(16) float  S[N_TOK][N_TOK + 1];
  __shared__ __align__(16) __bf16 P[N_TOK][N_TOK];
  __shared__ __align__(16) __bf16 Vs[N_TOK][HD_];

  const __bf16* base = QKV + (size_t)b * N_TOK * K3;
  const float ls = __expf(fminf(logit_scale[h], 4.6052f));

  // async-stage V rows into LDS (overlaps with QK^T and softmax; drained before P@V)
  for (int t = threadIdx.x; t < 256; t += 128) {          // 256 x 16B chunks
    int r = t >> 2;
    int d = (t & 3) * 8;
    wa2_async_b128(base + (size_t)r * K3 + 2 * CDIM + h * HD_ + d, &Vs[r][d]);
  }

  // ---- Q tile (A, 16x32), cosine-normalize row + fold logit scale
  V16 qa;
  {
    const __bf16* qrow = base + (size_t)(wave * 16 + l16) * K3 + h * HD_ + 16 * hh;
    qa.h[0] = *(const v8bf*)(qrow);
    qa.h[1] = *(const v8bf*)(qrow + 8);
    float ss = 0.f;
#pragma unroll
    for (int i = 0; i < 16; ++i) { float f = (float)qa.v[i]; ss = fmaf(f, f, ss); }
    ss += __shfl_xor(ss, 16, 32);               // lane pair L / L^16 holds the two row halves
    float sc = ls / fmaxf(sqrtf(ss), 1e-12f);
#pragma unroll
    for (int i = 0; i < 16; ++i) qa.v[i] = (__bf16)((float)qa.v[i] * sc);
  }

  // ---- S strip = (ls*qn) @ kn^T : 4 WMMAs
  v8f sacc[4] = {};
#pragma unroll
  for (int t = 0; t < 4; ++t) {
    const __bf16* krow = base + (size_t)(t * 16 + l16) * K3 + CDIM + h * HD_ + 16 * hh;
    V16 kb;                                     // B-tile column = one k row (contiguous)
    kb.h[0] = *(const v8bf*)(krow);
    kb.h[1] = *(const v8bf*)(krow + 8);
    float ss = 0.f;
#pragma unroll
    for (int i = 0; i < 16; ++i) { float f = (float)kb.v[i]; ss = fmaf(f, f, ss); }
    ss += __shfl_xor(ss, 16, 32);
    float sc = 1.0f / fmaxf(sqrtf(ss), 1e-12f);
#pragma unroll
    for (int i = 0; i < 16; ++i) kb.v[i] = (__bf16)((float)kb.v[i] * sc);
    sacc[t] = __builtin_amdgcn_wmma_f32_16x16x32_bf16(false, qa.v, false, kb.v,
                                                      (short)0, sacc[t], false, false);
  }
#pragma unroll
  for (int t = 0; t < 4; ++t)
#pragma unroll
    for (int j = 0; j < 8; ++j)
      S[wave * 16 + 8 * hh + j][t * 16 + l16] = sacc[t][j];
  __syncthreads();

  // ---- softmax rows, + bias + 2*mask (threads 0..63, one row each)
  if (threadIdx.x < 64) {
    const int r = threadIdx.x;
    const float* brow = bias + ((size_t)h * 64 + r) * 64;
    const float* mrow = mask + ((size_t)(b & 63) * 64 + r) * 64;
    float vals[64];
    float mx = -1e30f;
#pragma unroll
    for (int c = 0; c < 64; ++c) {
      float v = S[r][c] + brow[c] + 2.0f * mrow[c];
      vals[c] = v;
      mx = fmaxf(mx, v);
    }
    float sum = 0.f;
#pragma unroll
    for (int c = 0; c < 64; ++c) { float e = __expf(vals[c] - mx); vals[c] = e; sum += e; }
    float inv = 1.0f / sum;
#pragma unroll
    for (int c = 0; c < 64; ++c) P[r][c] = (__bf16)(vals[c] * inv);
  }
  asm volatile("s_wait_asynccnt 0x0" ::: "memory");       // V slab resident before P@V
  __syncthreads();

  // ---- O strip = P @ V : K=64 -> 2 K-steps, N=32 -> 2 tiles, 4 WMMAs
  v8f oacc[2] = {};
#pragma unroll
  for (int ks = 0; ks < 2; ++ks) {
    V16 pa;
    const __bf16* prow = &P[wave * 16 + l16][ks * 32 + 16 * hh];
    pa.h[0] = *(const v8bf*)(prow);
    pa.h[1] = *(const v8bf*)(prow + 8);
#pragma unroll
    for (int nt = 0; nt < 2; ++nt) {
      V16 vb;                                   // B-tile: lane col d = nt*16+l16, K = ks*32+16*hh+i
#pragma unroll
      for (int i = 0; i < 16; ++i)
        vb.v[i] = Vs[ks * 32 + 16 * hh + i][nt * 16 + l16];
      oacc[nt] = __builtin_amdgcn_wmma_f32_16x16x32_bf16(false, pa.v, false, vb.v,
                                                         (short)0, oacc[nt], false, false);
    }
  }
#pragma unroll
  for (int nt = 0; nt < 2; ++nt)
#pragma unroll
    for (int j = 0; j < 8; ++j) {
      int row = wave * 16 + 8 * hh + j;
      int d   = nt * 16 + l16;
      out[((size_t)b * N_TOK + row) * CDIM + h * HD_ + d] = oacc[nt][j];
    }
}

extern "C" void kernel_launch(void* const* d_in, const int* in_sizes, int n_in,
                              void* d_out, int out_size, void* d_ws, size_t ws_size,
                              hipStream_t stream) {
  const float* x    = (const float*)d_in[0];
  const float* mask = (const float*)d_in[1];
  const float* qkvw = (const float*)d_in[2];
  const float* qkvb = (const float*)d_in[3];
  const float* w1   = (const float*)d_in[4];
  const float* b1   = (const float*)d_in[5];
  const float* w2   = (const float*)d_in[6];
  const float* lsc  = (const float*)d_in[7];
  float* out = (float*)d_out;

  char* ws = (char*)d_ws;
  size_t off = 0;
  __bf16* x_bf = (__bf16*)(ws + off); off += (size_t)65536 * CDIM * 2;   // 64 MB
  __bf16* w_bf = (__bf16*)(ws + off); off += (size_t)K3 * CDIM * 2;      // 1.5 MB
  __bf16* qkv  = (__bf16*)(ws + off); off += (size_t)65536 * K3 * 2;     // 192 MB
  float*  tab  = (float*)(ws + off);  off += 225 * HEADS * 4; off = (off + 255) & ~(size_t)255;
  float*  biasB= (float*)(ws + off);  off += (size_t)HEADS * N_TOK * N_TOK * 4;

  wa2_cvt_bf16<<<(65536 * CDIM) / 256, 256, 0, stream>>>(x, x_bf, 65536 * CDIM);
  wa2_cvt_bf16<<<(K3 * CDIM) / 256, 256, 0, stream>>>(qkvw, w_bf, K3 * CDIM);
  wa2_cpb<<<225, 16, 0, stream>>>(w1, b1, w2, tab);
  wa2_bias<<<(HEADS * N_TOK * N_TOK) / 256, 256, 0, stream>>>(tab, biasB);
  wa2_qkv_gemm<<<dim3(65536 / 128, K3 / 64), 128, 0, stream>>>(x_bf, w_bf, qkvb, qkv);
  wa2_attn<<<dim3(NBATCH, HEADS), 128, 0, stream>>>(qkv, mask, biasB, lsc, out);
}
